// MentionScore_46651934769572
// MI455X (gfx1250) — matrix operand order repaired and checked by reading, not compiled, not run on hardware
//
#include <hip/hip_runtime.h>
#include <hip/hip_bf16.h>
#include <cstdint>

// ---------------- problem constants ----------------
#define T_TOK 2048
#define D_DIM 1024
#define H_DIM 1024
#define S_SPAN 20480
#define D3 3072
#define KPRUNE 819
#define NCHUNK 8          // 2048 / 256 row chunks for column scan
#define CHUNK 256

// ---------------- vector types ----------------
typedef __attribute__((ext_vector_type(8)))  unsigned short v8u;
typedef __attribute__((ext_vector_type(16))) unsigned short v16u;
typedef __attribute__((ext_vector_type(16))) __bf16        v16bf;
typedef __attribute__((ext_vector_type(8)))  float          v8f;

typedef unsigned short u16;
typedef unsigned int   u32;
typedef unsigned long long u64;

// ---------------- helpers ----------------
__device__ __forceinline__ u16 f2bf(float f) {
    u32 u = __float_as_uint(f);
    u32 r = u + 0x7FFFu + ((u >> 16) & 1u);   // round-to-nearest-even
    return (u16)(r >> 16);
}
__device__ __forceinline__ float bf2f(u16 b) {
    return __uint_as_float(((u32)b) << 16);
}
__device__ __forceinline__ u32 keyOf(float f) { // monotonic f32 -> u32 (ascending)
    u32 u = __float_as_uint(f);
    return (u & 0x80000000u) ? ~u : (u | 0x80000000u);
}

// CDNA5 async global->LDS copy (ASYNCcnt-tracked, bypasses VGPRs)
__device__ __forceinline__ void async_ld_b128(u32 ldsaddr, const void* g) {
    asm volatile("global_load_async_to_lds_b128 %0, %1, off"
                 :: "v"(ldsaddr), "v"((u64)(uintptr_t)g) : "memory");
}
__device__ __forceinline__ void wait_async0() {
    asm volatile("s_wait_asynccnt 0" ::: "memory");
}

// ---------------- f32 -> bf16 convert (row-major copy) ----------------
__global__ __launch_bounds__(256)
void cvt_f32_bf16(const float* __restrict__ in, u16* __restrict__ out, int n) {
    int i = blockIdx.x * 256 + threadIdx.x;
    if (i < n) out[i] = f2bf(in[i]);
}

// ---------------- f32 [K,N] -> bf16 transposed [N,K] (weights) ----------------
__global__ __launch_bounds__(256)
void cvt_transpose_bf16(const float* __restrict__ in, u16* __restrict__ out,
                        int K, int N) {
    __shared__ float tile[32][33];
    int tx = threadIdx.x & 31, ty = threadIdx.x >> 5;   // 32 x 8
    int n0 = blockIdx.x * 32, k0 = blockIdx.y * 32;
#pragma unroll
    for (int i = 0; i < 32; i += 8)
        tile[ty + i][tx] = in[(size_t)(k0 + ty + i) * N + n0 + tx];
    __syncthreads();
#pragma unroll
    for (int i = 0; i < 32; i += 8)
        out[(size_t)(n0 + ty + i) * K + k0 + tx] = f2bf(tile[tx][ty + i]);
}

// ---------------- bf16 WMMA GEMM: C = act(A[M,K] @ Bt[N,K]^T + bias) ----------------
// block tile 128x128, 8 waves (2 M x 4 N), each wave: 4x2 tiles of 16x16, K-step 32
// double-buffered LDS, async global->LDS staging, padded stride 40 (bank-conflict free)
#define BM 128
#define BN 128
#define KT 32
#define LSTR 40                 // padded LDS row stride (u16 elements)
#define ATILE (BM * LSTR)       // 5120 u16
#define LBUF  (2 * ATILE)       // A + B per buffer = 10240 u16

__global__ __launch_bounds__(256)
void gemm_bf16_wmma(const u16* __restrict__ A, const u16* __restrict__ Bt,
                    const float* __restrict__ bias,
                    float* __restrict__ outF, u16* __restrict__ outB,
                    int M, int N, int K, int doRelu)
{
    __shared__ __align__(16) u16 lds[2 * LBUF];   // 40 KB

    const int tid   = threadIdx.x;
    const int lane  = tid & 31;
    const int wid   = tid >> 5;
    const int waveM = wid & 1;              // 0..1
    const int waveN = wid >> 1;             // 0..3
    const int m0 = blockIdx.y * BM;
    const int n0 = blockIdx.x * BN;

    v8f acc[4][2];
#pragma unroll
    for (int i = 0; i < 4; ++i)
#pragma unroll
        for (int j = 0; j < 2; ++j)
#pragma unroll
            for (int e = 0; e < 8; ++e) acc[i][j][e] = 0.0f;

    // staging coordinates: thread -> (row 0..127, K-half 0/16), 2x b128 per matrix
    const int sRow  = tid >> 1;
    const int sHalf = (tid & 1) * 16;

    const u32 ldsBase = (u32)(uintptr_t)&lds[0];
    u32 aOff[2], bOff[2];
#pragma unroll
    for (int b = 0; b < 2; ++b) {
        aOff[b] = ldsBase + (u32)(b * LBUF + sRow * LSTR + sHalf) * 2;
        bOff[b] = ldsBase + (u32)(b * LBUF + ATILE + sRow * LSTR + sHalf) * 2;
    }
    const u16* gA = A  + (size_t)(m0 + sRow) * K + sHalf;
    const u16* gB = Bt + (size_t)(n0 + sRow) * K + sHalf;

    // prologue: stage K-tile 0 into buffer 0
    async_ld_b128(aOff[0],      gA);
    async_ld_b128(aOff[0] + 16, gA + 8);
    async_ld_b128(bOff[0],      gB);
    async_ld_b128(bOff[0] + 16, gB + 8);
    wait_async0();
    __syncthreads();

    const int laneM = lane & 15;
    const int kSel  = (lane < 16) ? 0 : 1;
    const int nk = K / KT;

    for (int k = 0; k < nk; ++k) {
        const int cur = k & 1, nxt = cur ^ 1;
        if (k + 1 < nk) {   // issue next tile's async copies (overlap with WMMA)
            const u16* ga = gA + (size_t)(k + 1) * KT;
            const u16* gb = gB + (size_t)(k + 1) * KT;
            async_ld_b128(aOff[nxt],      ga);
            async_ld_b128(aOff[nxt] + 16, ga + 8);
            async_ld_b128(bOff[nxt],      gb);
            async_ld_b128(bOff[nxt] + 16, gb + 8);
        }

        const u16* bufA = &lds[cur * LBUF];
        const u16* bufB = &lds[cur * LBUF + ATILE];

        v16bf af[4], bf[2];
#pragma unroll
        for (int i = 0; i < 4; ++i) {
            int r = waveM * 64 + i * 16 + laneM;
            v8u lo = *(const v8u*)&bufA[r * LSTR + kSel * 8];        // K kb..kb+7
            v8u hi = *(const v8u*)&bufA[r * LSTR + kSel * 8 + 16];   // K kb+16..kb+23
            v16u t = __builtin_shufflevector(lo, hi, 0,1,2,3,4,5,6,7,8,9,10,11,12,13,14,15);
            af[i] = __builtin_bit_cast(v16bf, t);
        }
#pragma unroll
        for (int j = 0; j < 2; ++j) {
            int c = waveN * 32 + j * 16 + laneM;
            v8u lo = *(const v8u*)&bufB[c * LSTR + kSel * 8];
            v8u hi = *(const v8u*)&bufB[c * LSTR + kSel * 8 + 16];
            v16u t = __builtin_shufflevector(lo, hi, 0,1,2,3,4,5,6,7,8,9,10,11,12,13,14,15);
            bf[j] = __builtin_bit_cast(v16bf, t);
        }
#pragma unroll
        for (int i = 0; i < 4; ++i)
#pragma unroll
            for (int j = 0; j < 2; ++j)
                acc[i][j] = __builtin_amdgcn_wmma_f32_16x16x32_bf16(
                    false, af[i], false, bf[j], (short)0, acc[i][j], false, false);

        if (k + 1 < nk) {
            wait_async0();       // this wave's next-tile copies landed in LDS
            __syncthreads();     // all waves done reading cur + writing nxt
        }
    }

    // epilogue: bias + relu, f32 and/or bf16 stores
    const int mHi = (lane < 16) ? 0 : 8;
#pragma unroll
    for (int j = 0; j < 2; ++j) {
        int n = n0 + waveN * 32 + j * 16 + laneM;
        float bv = bias ? bias[n] : 0.0f;
#pragma unroll
        for (int i = 0; i < 4; ++i) {
            int mb = m0 + waveM * 64 + i * 16 + mHi;
#pragma unroll
            for (int v = 0; v < 8; ++v) {
                float x = acc[i][j][v] + bv;
                if (doRelu) x = fmaxf(x, 0.0f);
                size_t off = (size_t)(mb + v) * N + n;
                if (outF) outF[off] = x;
                if (outB) outB[off] = f2bf(x);
            }
        }
    }
}

// ---------------- softmax over D per token, times embeds ----------------
__global__ __launch_bounds__(256)
void softmax_mul(const float* __restrict__ alpha, const float* __restrict__ emb,
                 float* __restrict__ weighted)
{
    __shared__ float red[256];
    int t = blockIdx.x, tid = threadIdx.x;
    const float* a = alpha + (size_t)t * D_DIM;
    float v[4];
    float mx = -INFINITY;
#pragma unroll
    for (int i = 0; i < 4; ++i) { v[i] = a[tid + 256 * i]; mx = fmaxf(mx, v[i]); }
    red[tid] = mx; __syncthreads();
    for (int s = 128; s > 0; s >>= 1) { if (tid < s) red[tid] = fmaxf(red[tid], red[tid + s]); __syncthreads(); }
    mx = red[0]; __syncthreads();
    float sum = 0.0f;
#pragma unroll
    for (int i = 0; i < 4; ++i) { v[i] = expf(v[i] - mx); sum += v[i]; }
    red[tid] = sum; __syncthreads();
    for (int s = 128; s > 0; s >>= 1) { if (tid < s) red[tid] += red[tid + s]; __syncthreads(); }
    float inv = 1.0f / red[0];
    const float* e = emb + (size_t)t * D_DIM;
    float* w = weighted + (size_t)t * D_DIM;
#pragma unroll
    for (int i = 0; i < 4; ++i) w[tid + 256 * i] = v[i] * inv * e[tid + 256 * i];
}

// ---------------- column-wise prefix sum (2-pass over 8 chunks) ----------------
__global__ __launch_bounds__(256)
void col_chunk_sum(const float* __restrict__ w, float* __restrict__ chunkSum) {
    int g = blockIdx.x * 256 + threadIdx.x;   // 0..8191
    int col = g & (D_DIM - 1), chunk = g >> 10;
    float s = 0.0f;
    for (int r = 0; r < CHUNK; ++r) s += w[(size_t)(chunk * CHUNK + r) * D_DIM + col];
    chunkSum[chunk * D_DIM + col] = s;
}
__global__ __launch_bounds__(256)
void col_chunk_off(float* __restrict__ chunkSum) {   // in-place exclusive scan over chunks
    int col = blockIdx.x * 256 + threadIdx.x;        // 0..1023
    float run = 0.0f;
    for (int c = 0; c < NCHUNK; ++c) {
        float v = chunkSum[c * D_DIM + col];
        chunkSum[c * D_DIM + col] = run;
        run += v;
    }
}
__global__ __launch_bounds__(256)
void col_prefix(const float* __restrict__ w, const float* __restrict__ chunkOff,
                float* __restrict__ prefix) {        // prefix[T+1][D]
    int g = blockIdx.x * 256 + threadIdx.x;
    int col = g & (D_DIM - 1), chunk = g >> 10;
    float run = chunkOff[chunk * D_DIM + col];
    if (chunk == 0) prefix[col] = 0.0f;
    for (int r = 0; r < CHUNK; ++r) {
        int t = chunk * CHUNK + r;
        run += w[(size_t)t * D_DIM + col];
        prefix[(size_t)(t + 1) * D_DIM + col] = run;
    }
}

// ---------------- build g = [lstm[start], lstm[end], prefix[end+1]-prefix[start]] ----------------
__global__ __launch_bounds__(256)
void build_g(const float* __restrict__ lstm, const float* __restrict__ prefix,
             const int* __restrict__ starts, const int* __restrict__ ends,
             float* __restrict__ gOut, u16* __restrict__ gBf)
{
    int s = blockIdx.x, tid = threadIdx.x;
    int st = starts[s], en = ends[s];
    const float* rs = lstm + (size_t)st * D_DIM;
    const float* re = lstm + (size_t)en * D_DIM;
    const float* p0 = prefix + (size_t)st * D_DIM;
    const float* p1 = prefix + (size_t)(en + 1) * D_DIM;
    size_t base = (size_t)s * D3;
    for (int d = tid; d < D_DIM; d += 256) {
        float a = rs[d], b = re[d], x = p1[d] - p0[d];
        gOut[base + d] = a;            gBf[base + d] = f2bf(a);
        gOut[base + D_DIM + d] = b;    gBf[base + D_DIM + d] = f2bf(b);
        gOut[base + 2*D_DIM + d] = x;  gBf[base + 2*D_DIM + d] = f2bf(x);
    }
}

// ---------------- GEMV: scores = s2 @ Ws3 + bs3 (one wave per row) ----------------
__global__ __launch_bounds__(256)
void gemv_ws3(const u16* __restrict__ s2, const float* __restrict__ Ws3,
              const float* __restrict__ bs3, float* __restrict__ scores)
{
    int lane = threadIdx.x & 31, wave = threadIdx.x >> 5;
    int row = blockIdx.x * 8 + wave;
    const u16* r = s2 + (size_t)row * H_DIM;
    float acc = 0.0f;
    for (int k = lane; k < H_DIM; k += 32) acc += bf2f(r[k]) * Ws3[k];
#pragma unroll
    for (int o = 16; o > 0; o >>= 1) acc += __shfl_down(acc, o, 32);
    if (lane == 0) scores[row] = acc + bs3[0];
}

// ---------------- exact top-k: radix-select threshold + bitonic sort ----------------
__global__ __launch_bounds__(1024)
void topk_select(const float* __restrict__ scores, int* __restrict__ outIdx)
{
    __shared__ int  hist[256];
    __shared__ u32  sh_byte;
    __shared__ int  sh_rem;
    __shared__ int  cnt;
    __shared__ u32  selU[1024];
    __shared__ int  selI[1024];
    int tid = threadIdx.x;

    u32 fixedMask = 0, fixedVal = 0;
    int rem = KPRUNE;
    for (int shift = 24; shift >= 0; shift -= 8) {
        if (tid < 256) hist[tid] = 0;
        __syncthreads();
        for (int i = tid; i < S_SPAN; i += 1024) {
            u32 u = keyOf(scores[i]);
            if ((u & fixedMask) == fixedVal) atomicAdd(&hist[(u >> shift) & 255], 1);
        }
        __syncthreads();
        if (tid == 0) {
            int acc = 0, b = 255;
            for (; b >= 0; --b) { int h = hist[b]; if (acc + h >= rem) break; acc += h; }
            if (b < 0) b = 0;
            sh_byte = (u32)b; sh_rem = rem - acc;
        }
        __syncthreads();
        fixedVal |= (sh_byte << shift);
        fixedMask |= (255u << shift);
        rem = sh_rem;
        __syncthreads();
    }
    u32 thr = fixedVal;   // exact key of the K-th largest

    if (tid == 0) cnt = 0;
    __syncthreads();
    for (int i = tid; i < S_SPAN; i += 1024) {
        u32 u = keyOf(scores[i]);
        if (u >= thr) {
            int p = atomicAdd(&cnt, 1);
            if (p < 1024) { selU[p] = u; selI[p] = i; }
        }
    }
    __syncthreads();
    int total = cnt;
    if (tid >= total) { selU[tid] = 0u; selI[tid] = 0x7fffffff; }
    __syncthreads();

    // bitonic sort: descending by key, ascending by index on ties (jax top_k order)
    for (int k = 2; k <= 1024; k <<= 1) {
        for (int j = k >> 1; j > 0; j >>= 1) {
            int ixj = tid ^ j;
            if (ixj > tid) {
                u32 ua = selU[tid], ub = selU[ixj];
                int ia = selI[tid], ib = selI[ixj];
                bool aAfterB  = (ua < ub) || (ua == ub && ia > ib);
                bool aBeforeB = (ua > ub) || (ua == ub && ia < ib);
                bool ascBlock = ((tid & k) == 0);
                bool doSwap = ascBlock ? aAfterB : aBeforeB;
                if (doSwap) { selU[tid] = ub; selU[ixj] = ua; selI[tid] = ib; selI[ixj] = ia; }
            }
            __syncthreads();
        }
    }
    if (tid < KPRUNE) outIdx[tid] = selI[tid];
}

// ---------------- host-side orchestration ----------------
static inline size_t alignUp256(size_t x) { return (x + 255) & ~(size_t)255; }

extern "C" void kernel_launch(void* const* d_in, const int* in_sizes, int n_in,
                              void* d_out, int out_size, void* d_ws, size_t ws_size,
                              hipStream_t stream)
{
    (void)in_sizes; (void)n_in; (void)out_size; (void)ws_size;
    const float* lstm = (const float*)d_in[0];
    const float* emb  = (const float*)d_in[1];
    const float* Wa1  = (const float*)d_in[2];  const float* ba1 = (const float*)d_in[3];
    const float* Wa2  = (const float*)d_in[4];  const float* ba2 = (const float*)d_in[5];
    const float* Wa3  = (const float*)d_in[6];  const float* ba3 = (const float*)d_in[7];
    const float* Ws1  = (const float*)d_in[8];  const float* bs1 = (const float*)d_in[9];
    const float* Ws2  = (const float*)d_in[10]; const float* bs2 = (const float*)d_in[11];
    const float* Ws3  = (const float*)d_in[12]; const float* bs3 = (const float*)d_in[13];
    const int* starts = (const int*)d_in[14];
    const int* ends   = (const int*)d_in[15];

    float* scores = (float*)d_out;                       // [S]
    float* gOut   = scores + S_SPAN;                     // [S, 3D]
    int*   idxOut = (int*)(gOut + (size_t)S_SPAN * D3);  // [KPRUNE]

    // workspace carve
    char* w = (char*)d_ws; size_t off = 0;
    auto carve = [&](size_t bytes) -> void* { void* p = w + off; off = alignUp256(off + bytes); return p; };
    u16* lstmB = (u16*)carve((size_t)T_TOK * D_DIM * 2);
    u16* Wa1t  = (u16*)carve((size_t)D_DIM * D_DIM * 2);   // transposed [N,K] bf16
    u16* Wa2t  = (u16*)carve((size_t)D_DIM * D_DIM * 2);
    u16* Wa3t  = (u16*)carve((size_t)D_DIM * D_DIM * 2);
    u16* Ws1t  = (u16*)carve((size_t)H_DIM * D3 * 2);
    u16* Ws2t  = (u16*)carve((size_t)H_DIM * H_DIM * 2);
    u16* h1b   = (u16*)carve((size_t)T_TOK * D_DIM * 2);
    u16* h2b   = (u16*)carve((size_t)T_TOK * D_DIM * 2);
    float* alpha    = (float*)carve((size_t)T_TOK * D_DIM * 4);
    float* weighted = (float*)carve((size_t)T_TOK * D_DIM * 4);
    float* prefix   = (float*)carve((size_t)(T_TOK + 1) * D_DIM * 4);
    float* chunkSum = (float*)carve((size_t)NCHUNK * D_DIM * 4);
    u16* gBf  = (u16*)carve((size_t)S_SPAN * D3 * 2);
    u16* s1b  = (u16*)carve((size_t)S_SPAN * H_DIM * 2);
    u16* s2b  = (u16*)carve((size_t)S_SPAN * H_DIM * 2);

    // 1. precision conversion: lstm row-major; weights converted + transposed to [N,K]
    cvt_f32_bf16<<<(T_TOK * D_DIM + 255) / 256, 256, 0, stream>>>(lstm, lstmB, T_TOK * D_DIM);
    cvt_transpose_bf16<<<dim3(D_DIM / 32, D_DIM / 32), 256, 0, stream>>>(Wa1, Wa1t, D_DIM, D_DIM);
    cvt_transpose_bf16<<<dim3(D_DIM / 32, D_DIM / 32), 256, 0, stream>>>(Wa2, Wa2t, D_DIM, D_DIM);
    cvt_transpose_bf16<<<dim3(D_DIM / 32, D_DIM / 32), 256, 0, stream>>>(Wa3, Wa3t, D_DIM, D_DIM);
    cvt_transpose_bf16<<<dim3(H_DIM / 32, D3 / 32), 256, 0, stream>>>(Ws1, Ws1t, D3, H_DIM);
    cvt_transpose_bf16<<<dim3(H_DIM / 32, H_DIM / 32), 256, 0, stream>>>(Ws2, Ws2t, H_DIM, H_DIM);

    // 2. token MLP chain (bf16 WMMA, f32 accum)
    dim3 blk(256);
    dim3 gTok(D_DIM / BN, T_TOK / BM);
    gemm_bf16_wmma<<<gTok, blk, 0, stream>>>(lstmB, Wa1t, ba1, nullptr, h1b, T_TOK, D_DIM, D_DIM, 1);
    gemm_bf16_wmma<<<gTok, blk, 0, stream>>>(h1b,   Wa2t, ba2, nullptr, h2b, T_TOK, D_DIM, D_DIM, 1);
    gemm_bf16_wmma<<<gTok, blk, 0, stream>>>(h2b,   Wa3t, ba3, alpha, nullptr, T_TOK, D_DIM, D_DIM, 0);

    // 3. softmax(alpha, feature dim) * embeds  (f32)
    softmax_mul<<<T_TOK, 256, 0, stream>>>(alpha, emb, weighted);

    // 4. column prefix sums (f32, exact segment sums)
    col_chunk_sum<<<(NCHUNK * D_DIM) / 256, 256, 0, stream>>>(weighted, chunkSum);
    col_chunk_off<<<D_DIM / 256, 256, 0, stream>>>(chunkSum);
    col_prefix<<<(NCHUNK * D_DIM) / 256, 256, 0, stream>>>(weighted, chunkSum, prefix);

    // 5. gather span features: g (f32 output) + bf16 copy (L2-resident for GEMM)
    build_g<<<S_SPAN, 256, 0, stream>>>(lstm, prefix, starts, ends, gOut, gBf);

    // 6. span scoring MLP (the heavy GEMMs)
    dim3 gSpan(H_DIM / BN, S_SPAN / BM);
    gemm_bf16_wmma<<<gSpan, blk, 0, stream>>>(gBf, Ws1t, bs1, nullptr, s1b, S_SPAN, H_DIM, D3, 1);
    gemm_bf16_wmma<<<gSpan, blk, 0, stream>>>(s1b, Ws2t, bs2, nullptr, s2b, S_SPAN, H_DIM, H_DIM, 1);

    // 7. final GEMV -> mention_scores (f32)
    gemv_ws3<<<S_SPAN / 8, 256, 0, stream>>>(s2b, Ws3, bs3, scores);

    // 8. exact top-k indices
    topk_select<<<1, 1024, 0, stream>>>(scores, idxOut);
}
